// MultiQueryAttention_7662221656422
// MI455X (gfx1250) — matrix operand order repaired and checked by reading, not compiled, not run on hardware
//
#include <hip/hip_runtime.h>
#include <hip/hip_bf16.h>

// ---------------------------------------------------------------------------
// MQA forward on gfx1250 (MI455X, wave32) via v_wmma_f32_16x16x32_bf16.
// B=4, S=2048, EMB=1024, NH=16, HD=64.
//
// Roofline: ~105 GFLOP over ~140MB -> mem floor ~6us @23.3TB/s; bf16-WMMA
// compute ~40us -> compute-bound. bf16 inputs + f32 accumulation.
// GEMM B panels and attention K/V tiles are staged to LDS with
// global_load_async_to_lds_b128 (ASYNCcnt), double-buffered so DMA overlaps
// WMMA. Softmax row-sums via WMMA (all-ones B); row-max via DPP16 butterflies.
// ---------------------------------------------------------------------------

typedef __bf16 v16bf __attribute__((ext_vector_type(16)));
typedef unsigned short u16;
typedef u16   v16us __attribute__((ext_vector_type(16)));
typedef u16   v8us  __attribute__((ext_vector_type(8)));
typedef float v8f   __attribute__((ext_vector_type(8)));
typedef float v4f   __attribute__((ext_vector_type(4)));

__device__ __forceinline__ u16 f2bf(float f) {
  unsigned u = __builtin_bit_cast(unsigned, f);
  u += 0x7FFFu + ((u >> 16) & 1u);   // round-to-nearest-even
  return (u16)(u >> 16);
}

__device__ __forceinline__ v16bf as_bf(v16us v) {
  return __builtin_bit_cast(v16bf, v);
}

// D = A*B + C, bf16 in, f32 accum
__device__ __forceinline__ v8f wmma_bf16(v16us a, v16us b, v8f c) {
  return __builtin_amdgcn_wmma_f32_16x16x32_bf16(
      false, as_bf(a), false, as_bf(b), (short)0, c, false, false);
}

// A-fragment (16x32 MxK, row-major source): lane half h holds
// K = k0 + 8h + {0..7} (elems 0-7) and K = k0 + 16 + 8h + {0..7} (elems 8-15).
__device__ __forceinline__ v16us a_frag_bf16(const u16* rowptr, int k0, int h) {
  const u16* p = rowptr + k0 + h * 8;
  v8us lo = *(const v8us*)p;
  v8us hi = *(const v8us*)(p + 16);
  return __builtin_shufflevector(lo, hi, 0, 1, 2, 3, 4, 5, 6, 7,
                                 8, 9, 10, 11, 12, 13, 14, 15);
}

__device__ __forceinline__ v16us a_frag_f32(const float* rowptr, int k0, int h) {
  const float* p = rowptr + k0 + h * 8;
  v4f x0 = *(const v4f*)p,        x1 = *(const v4f*)(p + 4);
  v4f y0 = *(const v4f*)(p + 16), y1 = *(const v4f*)(p + 20);
  v16us f;
#pragma unroll
  for (int i = 0; i < 4; i++) {
    f[i]      = f2bf(x0[i]);
    f[i + 4]  = f2bf(x1[i]);
    f[i + 8]  = f2bf(y0[i]);
    f[i + 12] = f2bf(y1[i]);
  }
  return f;
}

// B-fragment (32x16 KxN) read from K-contiguous storage:
// lane half h holds K = k0 + 8h + {0..7} (elems 0-7) and K = k0 + 16 + 8h + {0..7} (elems 8-15).
__device__ __forceinline__ v16us b_frag(const u16* colptr, int k0, int h) {
  const u16* p = colptr + k0 + h * 8;
  v8us lo = *(const v8us*)p;
  v8us hi = *(const v8us*)(p + 16);
  return __builtin_shufflevector(lo, hi, 0, 1, 2, 3, 4, 5, 6, 7,
                                 8, 9, 10, 11, 12, 13, 14, 15);
}

// ---- DPP16 butterfly max over the 16-lane half -----------------------------
#define DPPMOV(x, ctrl)                                                       \
  __builtin_bit_cast(float, __builtin_amdgcn_update_dpp(                      \
      __builtin_bit_cast(int, (x)), __builtin_bit_cast(int, (x)), (ctrl),     \
      0xf, 0xf, true))

__device__ __forceinline__ float red_max16(float v) {
  v = fmaxf(v, DPPMOV(v, 0xB1));   // quad_perm(1,0,3,2)
  v = fmaxf(v, DPPMOV(v, 0x4E));   // quad_perm(2,3,0,1)
  v = fmaxf(v, DPPMOV(v, 0x141));  // row_half_mirror
  v = fmaxf(v, DPPMOV(v, 0x140));  // row_mirror
  return v;
}

// ---- async global->LDS copy (ASYNCcnt) -------------------------------------
__device__ __forceinline__ unsigned lds_off(const void* p) {
  return (unsigned)(size_t)p;  // LDS aperture: low 32 bits = LDS byte offset
}

__device__ __forceinline__ void async_copy16(unsigned lds_byte_off, const u16* gptr) {
  asm volatile("global_load_async_to_lds_b128 %0, %1, off"
               :: "v"(lds_byte_off), "v"((unsigned long long)(size_t)gptr)
               : "memory");
}

__device__ __forceinline__ void wait_async0() {
  asm volatile("s_wait_asynccnt 0x0" ::: "memory");
}

// ---------------------------------------------------------------------------
// Weight transpose + f32->bf16 : WT[n][k] = bf16(W[k][n])
// ---------------------------------------------------------------------------
__global__ void transpose_bf16_kernel(const float* __restrict__ W,
                                      u16* __restrict__ WT, int K, int N) {
  int idx = blockIdx.x * blockDim.x + threadIdx.x;
  if (idx >= N * K) return;
  int n = idx / K, k = idx % K;
  WT[idx] = f2bf(W[(size_t)k * N + n]);
}

// ---------------------------------------------------------------------------
// WMMA GEMM: Out = scale * (A[M,K] @ BT[N,K]^T + bias[N])
// 256 threads = 8 waves; wave w computes a 16(M) x (NT*16)(N) strip.
// B panel [NT*16 n][64 k] staged to LDS (async, double-buffered) and shared by
// all 8 waves; A fragments loaded per-wave from global (rows are unique).
// OUT_MODE: 0 = bf16 row-major, 1 = bf16 transposed Out[n][m], 2 = f32.
// ---------------------------------------------------------------------------
template <bool A_F32, int OUT_MODE, int NT>
__global__ __launch_bounds__(256)
void gemm_wmma_kernel(const void* __restrict__ A, int lda,
                      const u16* __restrict__ BT, int ldbt,
                      const float* __restrict__ bias,
                      void* __restrict__ Out, int ldo,
                      int K, float scale) {
  constexpr int KC = 64;                         // k-chunk per stage
  __shared__ __attribute__((aligned(32))) u16 bt[2][NT * 16][KC];

  const int t    = threadIdx.x;
  const int lane = t & 31;
  const int wave = t >> 5;
  const int h = lane >> 4;
  const int c = lane & 15;
  const int m0 = blockIdx.y * 128 + wave * 16;
  const int n0 = blockIdx.x * (NT * 16);

  // stage B panel rows n0..n0+NT*16-1, cols kb..kb+KC-1 (16B units)
  auto stage = [&](int buf, int kb) {
    constexpr int UNITS = NT * 16 * (KC / 8);    // 16B units per panel
#pragma unroll
    for (int u = t; u < UNITS; u += 256) {
      const int n  = u >> 3;                     // 8 units per 64-elem row
      const int kc = (u & 7) * 8;
      async_copy16(lds_off(&bt[buf][n][kc]),
                   BT + (size_t)(n0 + n) * ldbt + kb + kc);
    }
  };

  v8f acc[NT];
#pragma unroll
  for (int tt = 0; tt < NT; tt++) acc[tt] = {};

  const float* arpf = (const float*)A + (size_t)(m0 + c) * lda;
  const u16*   arpb = (const u16*)A + (size_t)(m0 + c) * lda;

  stage(0, 0);
  for (int kb = 0; kb < K; kb += KC) {
    const int buf = (kb / KC) & 1;
    wait_async0();        // this wave's panel DMA done
    __syncthreads();      // everyone's DMA done; prev-iter LDS reads retired
    if (kb + KC < K) stage(buf ^ 1, kb + KC);   // overlap next DMA w/ compute

#pragma unroll
    for (int kk = 0; kk < KC; kk += 32) {
      const v16us a = A_F32 ? a_frag_f32(arpf, kb + kk, h)
                            : a_frag_bf16(arpb, kb + kk, h);
#pragma unroll
      for (int tt = 0; tt < NT; tt++)
        acc[tt] = wmma_bf16(a, b_frag(&bt[buf][tt * 16 + c][0], kk, h), acc[tt]);
    }
  }

#pragma unroll
  for (int tt = 0; tt < NT; tt++) {
    const int n = n0 + tt * 16 + c;
    const float bval = bias[n];
#pragma unroll
    for (int i = 0; i < 8; i++) {
      const int m = m0 + i + 8 * h;   // C layout: row = vgpr + 8*(lane>>4)
      const float val = (acc[tt][i] + bval) * scale;
      if (OUT_MODE == 0)      ((u16*)Out)[(size_t)m * ldo + n] = f2bf(val);
      else if (OUT_MODE == 1) ((u16*)Out)[(size_t)n * ldo + m] = f2bf(val);
      else                    ((float*)Out)[(size_t)m * ldo + n] = val;
    }
  }
}

// ---------------------------------------------------------------------------
// Flash attention (MQA): block = 8 waves = 128 query rows of one (b, head).
// 64-key blocks staged to LDS (async, double-buffered). Qp pre-scaled by
// log2(e)/sqrt(HD); softmax in base-2. Row-sum via WMMA against all-ones B.
// ---------------------------------------------------------------------------
__global__ __launch_bounds__(256)
void mqa_attention_kernel(const u16* __restrict__ Qp,
                          const u16* __restrict__ Kp,
                          const u16* __restrict__ VpT,
                          u16* __restrict__ AO,
                          int S, int BS) {
  __shared__ __attribute__((aligned(32))) u16 kbuf[2][64][64];  // [key][d]
  __shared__ __attribute__((aligned(32))) u16 vbuf[2][64][64];  // [d][key]
  __shared__ __attribute__((aligned(16))) u16 ptile[8][16][64];

  const int t    = threadIdx.x;
  const int lane = t & 31;
  const int wave = t >> 5;
  const int h = lane >> 4;
  const int c = lane & 15;
  const int b  = blockIdx.x >> 4;    // NH = 16
  const int hd = blockIdx.x & 15;
  const int q0 = blockIdx.y * 128 + wave * 16;

  const u16* kbase = Kp + (size_t)b * S * 64;    // [S][64]
  const u16* vbase = VpT + (size_t)b * S;        // row stride BS, cols = keys

  // Q fragments (contraction over d=64 -> 2 fragments), loaded once
  const u16* qrow = Qp + (size_t)(b * S + q0 + c) * 1024 + hd * 64;
  const v16us qa0 = a_frag_bf16(qrow, 0, h);
  const v16us qa1 = a_frag_bf16(qrow, 32, h);

  v16us ones;
#pragma unroll
  for (int i = 0; i < 16; i++) ones[i] = 0x3F80;  // bf16(1.0)

  v8f o0 = {}, o1 = {}, o2 = {}, o3 = {}, osum = {};
  float rmax[8];
#pragma unroll
  for (int i = 0; i < 8; i++) rmax[i] = -3.0e38f;

  // stage a 64-key K tile (8KB) + V tile (8KB): 1024 x 16B units / 256 thr
  auto stage = [&](int buf, int j) {
#pragma unroll
    for (int u = t; u < 512; u += 256) {
      const int kr = u >> 3, kc = (u & 7) * 8;
      async_copy16(lds_off(&kbuf[buf][kr][kc]),
                   kbase + (size_t)(j + kr) * 64 + kc);
      const int vd = u >> 3, vc = (u & 7) * 8;
      async_copy16(lds_off(&vbuf[buf][vd][vc]),
                   vbase + (size_t)vd * BS + j + vc);
    }
  };

  stage(0, 0);

  for (int j = 0; j < S; j += 64) {
    const int cur = (j >> 6) & 1;
    wait_async0();
    __syncthreads();
    if (j + 64 < S) stage(cur ^ 1, j + 64);     // overlap next DMA w/ compute

    // ---- scores: S[16q x 64k] = Q @ K^T (4 C tiles, contraction d=64) ----
    v8f s0 = {}, s1 = {}, s2 = {}, s3 = {};
    s0 = wmma_bf16(qa0, b_frag(&kbuf[cur][c][0],      0, h), s0);
    s0 = wmma_bf16(qa1, b_frag(&kbuf[cur][c][0],      32, h), s0);
    s1 = wmma_bf16(qa0, b_frag(&kbuf[cur][16 + c][0], 0, h), s1);
    s1 = wmma_bf16(qa1, b_frag(&kbuf[cur][16 + c][0], 32, h), s1);
    s2 = wmma_bf16(qa0, b_frag(&kbuf[cur][32 + c][0], 0, h), s2);
    s2 = wmma_bf16(qa1, b_frag(&kbuf[cur][32 + c][0], 32, h), s2);
    s3 = wmma_bf16(qa0, b_frag(&kbuf[cur][48 + c][0], 0, h), s3);
    s3 = wmma_bf16(qa1, b_frag(&kbuf[cur][48 + c][0], 32, h), s3);

    // ---- online softmax: rows = i + 8h; max via DPP16 butterflies ----
#pragma unroll
    for (int i = 0; i < 8; i++) {
      const float bm =
          red_max16(fmaxf(fmaxf(s0[i], s1[i]), fmaxf(s2[i], s3[i])));
      const float nm = fmaxf(rmax[i], bm);
      const float corr = exp2f(rmax[i] - nm);
      rmax[i] = nm;
      o0[i] *= corr; o1[i] *= corr; o2[i] *= corr; o3[i] *= corr;
      osum[i] *= corr;
      const float p0 = exp2f(s0[i] - nm);
      const float p1 = exp2f(s1[i] - nm);
      const float p2 = exp2f(s2[i] - nm);
      const float p3 = exp2f(s3[i] - nm);
      // C-tile -> A-fragment relayout through LDS (same-wave only, no barrier)
      ptile[wave][i + 8 * h][c]      = f2bf(p0);
      ptile[wave][i + 8 * h][c + 16] = f2bf(p1);
      ptile[wave][i + 8 * h][c + 32] = f2bf(p2);
      ptile[wave][i + 8 * h][c + 48] = f2bf(p3);
    }

    // ---- O += P[16x64] @ V[64x64]; row-sum rides on B = ones ----
    const v16us pf0 = a_frag_bf16(&ptile[wave][c][0], 0, h);
    const v16us pf1 = a_frag_bf16(&ptile[wave][c][0], 32, h);
    o0 = wmma_bf16(pf0, b_frag(&vbuf[cur][c][0],      0, h), o0);
    o0 = wmma_bf16(pf1, b_frag(&vbuf[cur][c][0],      32, h), o0);
    o1 = wmma_bf16(pf0, b_frag(&vbuf[cur][16 + c][0], 0, h), o1);
    o1 = wmma_bf16(pf1, b_frag(&vbuf[cur][16 + c][0], 32, h), o1);
    o2 = wmma_bf16(pf0, b_frag(&vbuf[cur][32 + c][0], 0, h), o2);
    o2 = wmma_bf16(pf1, b_frag(&vbuf[cur][32 + c][0], 32, h), o2);
    o3 = wmma_bf16(pf0, b_frag(&vbuf[cur][48 + c][0], 0, h), o3);
    o3 = wmma_bf16(pf1, b_frag(&vbuf[cur][48 + c][0], 32, h), o3);
    osum = wmma_bf16(pf0, ones, osum);
    osum = wmma_bf16(pf1, ones, osum);
  }

#pragma unroll
  for (int i = 0; i < 8; i++) {
    const float inv = 1.0f / osum[i];
    const size_t ro = (size_t)(b * S + q0 + i + 8 * h) * 1024 + hd * 64 + c;
    AO[ro + 0]  = f2bf(o0[i] * inv);
    AO[ro + 16] = f2bf(o1[i] * inv);
    AO[ro + 32] = f2bf(o2[i] * inv);
    AO[ro + 48] = f2bf(o3[i] * inv);
  }
}

// ---------------------------------------------------------------------------
extern "C" void kernel_launch(void* const* d_in, const int* in_sizes, int n_in,
                              void* d_out, int out_size, void* d_ws, size_t ws_size,
                              hipStream_t stream) {
  const float* q  = (const float*)d_in[0];
  const float* k  = (const float*)d_in[1];
  const float* v  = (const float*)d_in[2];
  const float* Wq = (const float*)d_in[3];
  const float* bq = (const float*)d_in[4];
  const float* Wk = (const float*)d_in[5];
  const float* bk = (const float*)d_in[6];
  const float* Wv = (const float*)d_in[7];
  const float* bv = (const float*)d_in[8];
  const float* Wo = (const float*)d_in[9];
  const float* bo = (const float*)d_in[10];
  float* out = (float*)d_out;

  const int B = 4, S = 2048, E = 1024, HD = 64;
  const int BS = B * S;  // 8192

  // Workspace carve-up (bf16 buffers), ~40 MB total
  char* ws = (char*)d_ws;
  u16* WqT = (u16*)ws;  ws += (size_t)E * E * 2;    // [1024][1024]
  u16* WkT = (u16*)ws;  ws += (size_t)HD * E * 2;   // [64][1024]
  u16* WvT = (u16*)ws;  ws += (size_t)HD * E * 2;   // [64][1024]
  u16* WoT = (u16*)ws;  ws += (size_t)E * E * 2;    // [1024][1024]
  u16* Qp  = (u16*)ws;  ws += (size_t)BS * E * 2;   // [8192][1024] (pre-scaled)
  u16* Kp  = (u16*)ws;  ws += (size_t)BS * HD * 2;  // [8192][64]
  u16* VpT = (u16*)ws;  ws += (size_t)HD * BS * 2;  // [64][8192]
  u16* AO  = (u16*)ws;  ws += (size_t)BS * E * 2;   // [8192][1024]

  // 1) Weights -> bf16, transposed to [N][K]
  transpose_bf16_kernel<<<(E * E + 255) / 256, 256, 0, stream>>>(Wq, WqT, E, E);
  transpose_bf16_kernel<<<(HD * E + 255) / 256, 256, 0, stream>>>(Wk, WkT, E, HD);
  transpose_bf16_kernel<<<(HD * E + 255) / 256, 256, 0, stream>>>(Wv, WvT, E, HD);
  transpose_bf16_kernel<<<(E * E + 255) / 256, 256, 0, stream>>>(Wo, WoT, E, E);

  // 1/sqrt(HD) * log2(e): softmax computed with exp2 (v_exp_f32 is base-2)
  const float qscale = 0.125f * 1.44269504088896340736f;

  // 2) Projections (A = f32 activations converted in-register)
  gemm_wmma_kernel<true, 0, 8><<<dim3(E / 128, BS / 128), 256, 0, stream>>>(
      q, E, WqT, E, bq, Qp, E, E, qscale);
  gemm_wmma_kernel<true, 0, 4><<<dim3(1, BS / 128), 256, 0, stream>>>(
      k, E, WkT, E, bk, Kp, HD, E, 1.0f);
  gemm_wmma_kernel<true, 1, 4><<<dim3(1, BS / 128), 256, 0, stream>>>(
      v, E, WvT, E, bv, VpT, BS, E, 1.0f);   // store transposed [64][B*S]

  // 3) Flash attention: grid = (B*NH, S/128)
  mqa_attention_kernel<<<dim3(B * 16, S / 128), 256, 0, stream>>>(
      Qp, Kp, VpT, AO, S, BS);

  // 4) Output projection (bf16 A, f32 out + bias)
  gemm_wmma_kernel<false, 2, 8><<<dim3(E / 128, BS / 128), 256, 0, stream>>>(
      AO, E, WoT, E, bo, out, E, E, 1.0f);
}
